// mixprop_61564061221148
// MI455X (gfx1250) — matrix operand order repaired and checked by reading, not compiled, not run on hardware
//
#include <hip/hip_runtime.h>

typedef __attribute__((ext_vector_type(16))) __bf16 v16bf;
typedef __attribute__((ext_vector_type(8)))  __bf16 v8bf;
typedef __attribute__((ext_vector_type(2)))  __bf16 v2bf;
typedef __attribute__((ext_vector_type(8)))  float  v8f;

#define ALPHA_ 0.05f
#define N_    32
#define C_    64
#define V_    64
#define L_    512
#define LB    16      // l-tile per workgroup
#define CB    16      // channel chunk
#define STR   72      // LDS row stride in halves (144B -> 16B-aligned fragments)

// d_ws layout (halves): [0,4096) A1 ; [4096,8192) A2 ; [8192,8192+12288) W' [chunk][o][48]
#define WS_A1 0
#define WS_A2 4096
#define WS_W  8192

// f32 pair -> packed bf16 (hardware v_cvt_pk_bf16_f32)
__device__ __forceinline__ v2bf pk2(float a, float b) {
#if __has_builtin(__builtin_amdgcn_cvt_pk_bf16_f32)
  return __builtin_amdgcn_cvt_pk_bf16_f32(a, b);
#else
  v2bf r; r[0] = (__bf16)a; r[1] = (__bf16)b; return r;
#endif
}
__device__ __forceinline__ __bf16 bf1(float a) { return pk2(a, a)[0]; }

// A-operand (16-bit, 16x32, M=row, K contiguous in LDS row):
// lane<16: halves K{0..15}+kstep ; lane>=16: K{16..31}+kstep
__device__ __forceinline__ v16bf ldsA(const __bf16* row, int kstep, int lane) {
  int base = kstep + ((lane & 16) ? 16 : 0);
  v8bf lo = *(const v8bf*)(row + base);
  v8bf hi = *(const v8bf*)(row + base + 8);
  return __builtin_shufflevector(lo, hi, 0,1,2,3,4,5,6,7,8,9,10,11,12,13,14,15);
}
// B-operand (16-bit, 32x16): lane n=lane&15, 16 contiguous K halves, +16 if lane>=16
__device__ __forceinline__ v16bf ldsB(const __bf16* row, int kstep, int lane) {
  int base = kstep + ((lane & 16) ? 16 : 0);
  v8bf lo = *(const v8bf*)(row + base);
  v8bf hi = *(const v8bf*)(row + base + 8);
  return __builtin_shufflevector(lo, hi, 0,1,2,3,4,5,6,7,8,9,10,11,12,13,14,15);
}

// ---------------- prep kernel: one workgroup, shared across all 1024 main blocks --------
__global__ __launch_bounds__(512, 1)
void mixprop_prep_kernel(const float* __restrict__ adj, const float* __restrict__ W,
                         __bf16* __restrict__ ws) {
  __shared__ float sAf[V_ * V_];
  const int t = threadIdx.x;

  if (t < V_) {                              // row-normalized adjacency (f32)
    float d = 0.f;
    for (int w = 0; w < V_; ++w) d += adj[t * V_ + w] + (w == t ? 1.f : 0.f);
    float inv = 1.f / d;
    for (int w = 0; w < V_; ++w)
      sAf[t * V_ + w] = (adj[t * V_ + w] + (w == t ? 1.f : 0.f)) * inv;
  }
  __syncthreads();

  {                                          // A1 = (1-a)A+aI ; A2 = aI+(1-a)A@A1
    const int v  = t >> 3;
    const int wb = (t & 7) * 8;
    float s[8];
    #pragma unroll
    for (int j = 0; j < 8; ++j) s[j] = 0.f;
    for (int k = 0; k < V_; ++k) {
      float avk = sAf[v * V_ + k];
      #pragma unroll
      for (int j = 0; j < 8; ++j) s[j] += avk * sAf[k * V_ + wb + j];
    }
    #pragma unroll
    for (int j = 0; j < 8; j += 2) {
      float a1v[2], a2v[2];
      #pragma unroll
      for (int u = 0; u < 2; ++u) {
        int w = wb + j + u;
        float avw = sAf[v * V_ + w];
        float dia = (v == w) ? ALPHA_ : 0.f;
        a1v[u] = (1.f - ALPHA_) * avw + dia;
        float aa1 = (1.f - ALPHA_) * s[j + u] + ALPHA_ * avw;
        a2v[u] = dia + (1.f - ALPHA_) * aa1;
      }
      *(v2bf*)(ws + WS_A1 + v * V_ + wb + j) = pk2(a1v[0], a1v[1]);
      *(v2bf*)(ws + WS_A2 + v * V_ + wb + j) = pk2(a2v[0], a2v[1]);
    }
  }
  // W' relabeled: [chunk][o][kc<48]; kc<16 -> hop0(c=kc); kc>=16 -> c=(kc-16)>>1, hop=1+((kc-16)&1)
  for (int gi = t; gi < 4 * C_ * 64; gi += 512) {
    int ch = gi >> 12, o = (gi >> 6) & 63, kc = gi & 63;
    if (kc < 48) {
      int cl, hop;
      if (kc < 16) { cl = kc; hop = 0; }
      else         { cl = (kc - 16) >> 1; hop = 1 + ((kc - 16) & 1); }
      ws[WS_W + ch * (C_ * 48) + o * 48 + kc] = bf1(W[o * 192 + hop * 64 + ch * 16 + cl]);
    }
  }
}

// ---------------- main fused kernel ----------------
__global__ __launch_bounds__(512, 1)
void mixprop_fused_kernel(const float* __restrict__ x, const float* __restrict__ b,
                          const __bf16* __restrict__ ws, float* __restrict__ out) {
  // LDS (~213 KB of the 320 KB WGP budget)
  __shared__ __bf16 sA1[V_ * STR];       // A1 [v][w]
  __shared__ __bf16 sA2[V_ * STR];       // A2 [v][w]
  __shared__ __bf16 sX [CB * LB * STR];  // X chunk [c][l][w] (w-contiguous)
  // sK kc labeling (interleaved): kc=c -> X(c); kc=16+2c -> H1(c); kc=17+2c -> H2(c); 48..63 zero
  __shared__ __bf16 sK [V_ * LB * STR];
  __shared__ __bf16 sW [C_ * STR];       // W' [o][kc] (48 data + 16 zeros)
  __shared__ float  sb [C_];

  const int t    = threadIdx.x;
  const int lane = t & 31;
  const int wv   = t >> 5;            // wave 0..15
  const int ln   = lane & 15;
  const int hif  = (lane >> 4) & 1;
  const int n    = blockIdx.y;
  const int l0   = blockIdx.x * LB;

  // ---- phase 0: pull precomputed A1/A2 from ws (16B segments, re-padded to STR) ----
  #pragma unroll
  for (int r = 0; r < 2; ++r) {
    int i = r * 512 + t;               // 1024 8-half segments (512 per matrix)
    int m = i >> 9, seg = i & 511;
    int v = seg >> 3, wb = (seg & 7) * 8;
    const __bf16* src = ws + (m ? WS_A2 : WS_A1) + v * V_ + wb;
    __bf16* dst = (m ? sA2 : sA1) + v * STR + wb;
    *(v8bf*)dst = *(const v8bf*)src;
  }
  if (t < C_) sb[t] = b[t];
  // zero the K padding regions (only written here; chunks touch kc 0..47)
  for (int i = t; i < V_ * LB * 16; i += 512)
    sK[(i >> 4) * STR + 48 + (i & 15)] = (__bf16)0.f;
  for (int i = t; i < C_ * 16; i += 512)
    sW[(i >> 4) * STR + 48 + (i & 15)] = (__bf16)0.f;

  v8f acc2[16];
  #pragma unroll
  for (int i = 0; i < 16; ++i)
    #pragma unroll
    for (int g = 0; g < 8; ++g) acc2[i][g] = 0.f;

  // Step-B mapping: wave = (channel-group, v-tile)
  const int cg  = (wv & 3) * 4;
  const int vt  = wv >> 2;
  // Step-D mapping: wave = (o-tile pair, v-group of 8)
  const int opair = wv & 1;
  const int vbase = (wv >> 1) * 8;

  for (int chunk = 0; chunk < 4; ++chunk) {
    const int cbase = chunk * CB;
    __syncthreads();                       // previous chunk's readers done

    // ---- Step A: stage x chunk (f32 -> bf16) into both layouts; copy W' slab ----
    #pragma unroll
    for (int q = 0; q < 8; ++q) {
      int gi = q * 512 + t;                // 16c * 64v * 4 quads = 4096
      int c = gi >> 8, rem = gi & 255, v = rem >> 2, lq = rem & 3;
      size_t gidx = ((size_t)((n * C_ + cbase + c) * V_ + v)) * L_ + l0 + lq * 4;
      const float4 xv = *(const float4*)(x + gidx);
      if (chunk < 3)                       // pull next chunk toward L2
        __builtin_prefetch(x + gidx + (size_t)CB * V_ * L_, 0, 1);
      v2bf plo = pk2(xv.x, xv.y);
      v2bf phi = pk2(xv.z, xv.w);
      __bf16 xs[4] = {plo[0], plo[1], phi[0], phi[1]};
      #pragma unroll
      for (int j = 0; j < 4; ++j) {
        int l = lq * 4 + j;
        sX[(c * LB + l) * STR + v] = xs[j];          // w-contiguous
        sK[(v * LB + l) * STR + c] = xs[j];          // kc-contiguous (X region)
      }
    }
    if (t < 384) {                         // W' slab: 64 rows x 48 halves in 16B segments
      int o = t / 6, s6 = t - o * 6;
      *(v8bf*)(sW + o * STR + s6 * 8) =
          *(const v8bf*)(ws + WS_W + chunk * (C_ * 48) + o * 48 + s6 * 8);
    }
    __syncthreads();

    // ---- Step B: H1 = A1@X, H2 = A2@X; A fragments reused over 4 channels ----
    {
      const __bf16* a1row = sA1 + (vt * 16 + ln) * STR;
      const __bf16* a2row = sA2 + (vt * 16 + ln) * STR;
      v16bf a10 = ldsA(a1row, 0, lane),  a11 = ldsA(a1row, 32, lane);
      v16bf a20 = ldsA(a2row, 0, lane),  a21 = ldsA(a2row, 32, lane);
      #pragma unroll
      for (int ci = 0; ci < 4; ++ci) {
        int c = cg + ci;
        const __bf16* brow = sX + (c * LB + ln) * STR;
        v16bf b0 = ldsB(brow, 0, lane);
        v16bf b1 = ldsB(brow, 32, lane);
        v8f h1, h2;
        #pragma unroll
        for (int g = 0; g < 8; ++g) { h1[g] = 0.f; h2[g] = 0.f; }
        h1 = __builtin_amdgcn_wmma_f32_16x16x32_bf16(false, a10, false, b0, (short)0, h1, false, false);
        h1 = __builtin_amdgcn_wmma_f32_16x16x32_bf16(false, a11, false, b1, (short)0, h1, false, false);
        h2 = __builtin_amdgcn_wmma_f32_16x16x32_bf16(false, a20, false, b0, (short)0, h2, false, false);
        h2 = __builtin_amdgcn_wmma_f32_16x16x32_bf16(false, a21, false, b1, (short)0, h2, false, false);
        #pragma unroll
        for (int g = 0; g < 8; ++g) {
          int vm = vt * 16 + g + hif * 8;
          // H1/H2 at kc = 16+2c / 17+2c -> packed b32 store (merges to b128 over ci)
          *(v2bf*)(sK + (vm * LB + ln) * STR + 16 + 2 * c) = pk2(h1[g], h2[g]);
        }
      }
    }
    __syncthreads();

    // ---- Step D: Out += W' @ K2 ; each B fragment feeds 2 o-tiles ----
    #pragma unroll
    for (int ks = 0; ks < 2; ++ks) {
      const __bf16* wrow0 = sW + ((opair * 2 + 0) * 16 + ln) * STR;
      const __bf16* wrow1 = sW + ((opair * 2 + 1) * 16 + ln) * STR;
      v16bf aw0 = ldsA(wrow0, ks * 32, lane);
      v16bf aw1 = ldsA(wrow1, ks * 32, lane);
      #pragma unroll
      for (int vi = 0; vi < 8; ++vi) {
        const __bf16* krow = sK + ((vbase + vi) * LB + ln) * STR;
        v16bf bk = ldsB(krow, ks * 32, lane);
        acc2[vi]     = __builtin_amdgcn_wmma_f32_16x16x32_bf16(false, aw0, false, bk,
                                                               (short)0, acc2[vi], false, false);
        acc2[8 + vi] = __builtin_amdgcn_wmma_f32_16x16x32_bf16(false, aw1, false, bk,
                                                               (short)0, acc2[8 + vi], false, false);
      }
    }
  }

  // ---- epilogue: bias + store ----
  #pragma unroll
  for (int j = 0; j < 16; ++j) {
    int ot = opair * 2 + (j >> 3);
    int v  = vbase + (j & 7);
    #pragma unroll
    for (int g = 0; g < 8; ++g) {
      int o = ot * 16 + g + hif * 8;
      out[((size_t)((n * C_ + o) * V_ + v)) * L_ + l0 + ln] = acc2[j][g] + sb[o];
    }
  }
}

extern "C" void kernel_launch(void* const* d_in, const int* in_sizes, int n_in,
                              void* d_out, int out_size, void* d_ws, size_t ws_size,
                              hipStream_t stream) {
  (void)in_sizes; (void)n_in; (void)ws_size; (void)out_size;
  const float* x   = (const float*)d_in[0];
  const float* adj = (const float*)d_in[1];
  const float* W   = (const float*)d_in[2];
  const float* b   = (const float*)d_in[3];
  float* out = (float*)d_out;
  __bf16* ws = (__bf16*)d_ws;          // needs 40 KB scratch

  mixprop_prep_kernel<<<dim3(1), dim3(512), 0, stream>>>(adj, W, ws);
  dim3 grid(L_ / LB, N_);              // 32 l-tiles x 32 batches = 1024 workgroups
  mixprop_fused_kernel<<<grid, dim3(512), 0, stream>>>(x, b, ws, out);
}
